// ProbSparseAttention_17471926960804
// MI455X (gfx1250) — matrix operand order repaired and checked by reading, not compiled, not run on hardware
//
#include <hip/hip_runtime.h>
#include <math.h>

typedef __attribute__((ext_vector_type(16))) _Float16 v16h;
typedef __attribute__((ext_vector_type(8)))  float    v8f;

// Problem constants (from reference setup_inputs)
constexpr int BB = 4;
constexpr int HH = 8;
constexpr int LL = 2048;
constexpr int DD = 64;
constexpr int UU = 40;          // FACTOR * ceil(ln(2048)) = 40
constexpr int NT = 3;           // ceil(40/16) M-tiles
constexpr int MC = 16;          // row-chunks for the V column-mean
constexpr float SCALE = 0.125f; // 1/sqrt(64)

// ---------------------------------------------------------------------------
// K1: sparsity metric M[b,h,q] = max_s(q.K_sample) - sum_s(q.K_sample)/L
// ---------------------------------------------------------------------------
__global__ void k_metric(const float* __restrict__ Q, const float* __restrict__ K,
                         const int* __restrict__ samp, float* __restrict__ M)
{
    int t = blockIdx.x * blockDim.x + threadIdx.x;
    if (t >= BB * HH * LL) return;
    int q  = t % LL;
    int bh = t / LL;

    const float4* q4 = (const float4*)(Q + (size_t)t * DD);
    float4 qv[16];
    #pragma unroll
    for (int i = 0; i < 16; ++i) qv[i] = q4[i];

    float mx = -INFINITY, sm = 0.f;
    for (int s = 0; s < UU; ++s) {
        int ks = samp[q * UU + s];
        const float4* k4 = (const float4*)(K + ((size_t)bh * LL + ks) * DD);
        float acc = 0.f;
        #pragma unroll
        for (int i = 0; i < 16; ++i) {
            float4 kv = k4[i];
            acc += qv[i].x * kv.x + qv[i].y * kv.y + qv[i].z * kv.z + qv[i].w * kv.w;
        }
        mx = fmaxf(mx, acc);
        sm += acc;
    }
    M[t] = mx - sm * (1.0f / (float)LL);
}

// ---------------------------------------------------------------------------
// K2: per-head top-40 of M (iterative argmax; ties -> lowest index)
// ---------------------------------------------------------------------------
__global__ void k_topk(const float* __restrict__ M, int* __restrict__ topk)
{
    __shared__ float mv[LL];
    __shared__ float rv[256];
    __shared__ int   ri[256];
    int bh  = blockIdx.x;
    int tid = threadIdx.x;

    for (int i = tid; i < LL; i += 256) mv[i] = M[(size_t)bh * LL + i];
    __syncthreads();

    for (int u = 0; u < UU; ++u) {
        float best = -INFINITY; int bi = LL;
        for (int i = tid; i < LL; i += 256) {
            float v = mv[i];
            if (v > best) { best = v; bi = i; }
        }
        rv[tid] = best; ri[tid] = bi;
        __syncthreads();
        for (int s = 128; s > 0; s >>= 1) {
            if (tid < s) {
                if (rv[tid + s] > rv[tid] ||
                    (rv[tid + s] == rv[tid] && ri[tid + s] < ri[tid])) {
                    rv[tid] = rv[tid + s]; ri[tid] = ri[tid + s];
                }
            }
            __syncthreads();
        }
        if (tid == 0) {
            topk[bh * UU + u] = ri[0];
            mv[ri[0]] = -INFINITY;
        }
        __syncthreads();
    }
}

// ---------------------------------------------------------------------------
// K3: column means of V, two-stage for parallelism (512 blocks stream 16 MB)
// ---------------------------------------------------------------------------
__global__ void k_mean_part(const float* __restrict__ V, float* __restrict__ part)
{
    int blk = blockIdx.x;             // bh * MC + ch
    int bh  = blk / MC, ch = blk % MC;
    int d   = threadIdx.x & 63;
    int sr  = threadIdx.x >> 6;       // 0..3 sub-row groups
    const int rows = LL / MC;         // 128
    const float* vp = V + ((size_t)bh * LL + ch * rows) * DD + d;
    float s = 0.f;
    for (int i = sr; i < rows; i += 4) s += vp[(size_t)i * DD];
    __shared__ float red[4][64];
    red[sr][d] = s;
    __syncthreads();
    if (sr == 0)
        part[(size_t)blk * DD + d] = red[0][d] + red[1][d] + red[2][d] + red[3][d];
}

__global__ void k_mean_red(const float* __restrict__ part, float* __restrict__ mean)
{
    int bh = blockIdx.x, d = threadIdx.x;
    float s = 0.f;
    #pragma unroll
    for (int c = 0; c < MC; ++c) s += part[((size_t)bh * MC + c) * DD + d];
    mean[bh * DD + d] = s * (1.0f / (float)LL);
}

__global__ void k_fill(const float* __restrict__ mean, float* __restrict__ out)
{
    size_t i = (size_t)blockIdx.x * blockDim.x + threadIdx.x;
    if (i >= (size_t)BB * HH * LL * DD) return;
    int d  = (int)(i & (DD - 1));
    int bh = (int)(i / ((size_t)LL * DD));
    out[i] = mean[bh * DD + d];
}

// ---------------------------------------------------------------------------
// K4a: split-K WMMA flash attention partials. Grid = BH*KC blocks of 96
//      (3 waves = 3 M-tiles of 16 rows). Each block streams LL/KC keys,
//      32 per iteration, and emits un-normalized O plus running (m, l).
// ---------------------------------------------------------------------------
__global__ void __launch_bounds__(96)
k_attn_part(const float* __restrict__ Q, const float* __restrict__ K,
            const float* __restrict__ V, const int* __restrict__ topk,
            float* __restrict__ Opart, float* __restrict__ mpart,
            float* __restrict__ lpart, int KC)
{
    __shared__ float    Ssm[NT][16][32];
    __shared__ _Float16 Pld[NT][16][32];
    __shared__ float    Ald[NT][16];

    const int bh   = blockIdx.x / KC;
    const int ch   = blockIdx.x % KC;
    const int wave = threadIdx.x >> 5;
    const int lane = threadIdx.x & 31;
    const int m    = lane & 15;   // A: M row / B,C,D: N column
    const int half = lane >> 4;   // K-split half

    const int nkeys = LL / KC;
    const int j0    = ch * nkeys;

    const float* Qh = Q + (size_t)bh * LL * DD;
    const float* Kh = K + (size_t)bh * LL * DD;
    const float* Vh = V + (size_t)bh * LL * DD;
    const int*   tk = topk + bh * UU;

    // A = selected Q rows (16x64 f16, softmax scale folded), contiguous k = 16*half + s
    int row    = wave * 16 + m;
    int rclamp = row < UU ? row : UU - 1;      // pad rows 40..47 (never stored)
    const float* qp = Qh + (size_t)tk[rclamp] * DD;
    v16h aq0, aq1;
    #pragma unroll
    for (int p = 0; p < 8; ++p) {
        int kb = p * 2 + half * 16;
        aq0[2 * p]     = (_Float16)(qp[kb]          * SCALE);
        aq0[2 * p + 1] = (_Float16)(qp[kb + 1]      * SCALE);
        aq1[2 * p]     = (_Float16)(qp[32 + kb]     * SCALE);
        aq1[2 * p + 1] = (_Float16)(qp[32 + kb + 1] * SCALE);
    }

    v8f o0 = {}, o1 = {}, o2 = {}, o3 = {};
    float m_i = -INFINITY, l_i = 0.f;          // valid on lanes 0..15 (row = lane)

    for (int j = j0; j < j0 + nkeys; j += 32) {
        // ---- S tiles: 16 rows x 32 keys (K=64 split into two WMMAs each) ----
        v8f s0 = {}, s1 = {};
        {
            const float* kr = Kh + (size_t)(j + m) * DD + half * 16;
            v16h b0, b1;
            #pragma unroll
            for (int i = 0; i < 16; ++i) { b0[i] = (_Float16)kr[i]; b1[i] = (_Float16)kr[32 + i]; }
            s0 = __builtin_amdgcn_wmma_f32_16x16x32_f16(false, aq0, false, b0, (short)0, s0, false, false);
            s0 = __builtin_amdgcn_wmma_f32_16x16x32_f16(false, aq1, false, b1, (short)0, s0, false, false);
        }
        {
            const float* kr = Kh + (size_t)(j + 16 + m) * DD + half * 16;
            v16h b0, b1;
            #pragma unroll
            for (int i = 0; i < 16; ++i) { b0[i] = (_Float16)kr[i]; b1[i] = (_Float16)kr[32 + i]; }
            s1 = __builtin_amdgcn_wmma_f32_16x16x32_f16(false, aq0, false, b0, (short)0, s1, false, false);
            s1 = __builtin_amdgcn_wmma_f32_16x16x32_f16(false, aq1, false, b1, (short)0, s1, false, false);
        }
        // C/D layout: lane holds column N=m, rows M = 8*half + r
        #pragma unroll
        for (int r = 0; r < 8; ++r) {
            Ssm[wave][half * 8 + r][m]      = s0[r];
            Ssm[wave][half * 8 + r][16 + m] = s1[r];
        }
        __syncthreads();

        // ---- online softmax: lanes 0..15 each own one row ----
        if (lane < 16) {
            float rmax = -INFINITY;
            #pragma unroll
            for (int c = 0; c < 32; ++c) rmax = fmaxf(rmax, Ssm[wave][lane][c]);
            float mnew  = fmaxf(m_i, rmax);
            float alpha = __expf(m_i - mnew);
            float ls = 0.f;
            #pragma unroll
            for (int c = 0; c < 32; ++c) {
                float p = __expf(Ssm[wave][lane][c] - mnew);
                Pld[wave][lane][c] = (_Float16)p;
                ls += p;
            }
            l_i = l_i * alpha + ls;
            m_i = mnew;
            Ald[wave][lane] = alpha;
        }
        __syncthreads();

        // ---- rescale O, then O += P(16x32) . V(32x64) ----
        #pragma unroll
        for (int r = 0; r < 8; ++r) {
            float a = Ald[wave][half * 8 + r];
            o0[r] *= a; o1[r] *= a; o2[r] *= a; o3[r] *= a;
        }
        v16h ap;
        #pragma unroll
        for (int p = 0; p < 8; ++p) {
            int kb = p * 2 + half * 16;
            ap[2 * p]     = Pld[wave][m][kb];
            ap[2 * p + 1] = Pld[wave][m][kb + 1];
        }
        v8f oacc[4] = { o0, o1, o2, o3 };
        #pragma unroll
        for (int t = 0; t < 4; ++t) {
            v16h bv;
            #pragma unroll
            for (int i = 0; i < 16; ++i)
                bv[i] = (_Float16)Vh[(size_t)(j + half * 16 + i) * DD + t * 16 + m];
            oacc[t] = __builtin_amdgcn_wmma_f32_16x16x32_f16(false, ap, false, bv, (short)0, oacc[t], false, false);
        }
        o0 = oacc[0]; o1 = oacc[1]; o2 = oacc[2]; o3 = oacc[3];
    }

    // ---- emit partials (un-normalized O, running m and l) ----
    const size_t pc = (size_t)bh * KC + ch;
    if (lane < 16) {
        int rr = wave * 16 + lane;
        if (rr < UU) {
            mpart[pc * UU + rr] = m_i;
            lpart[pc * UU + rr] = l_i;
        }
    }
    #pragma unroll
    for (int r = 0; r < 8; ++r) {
        int rrow = wave * 16 + half * 8 + r;
        if (rrow < UU) {
            float* op = Opart + (pc * UU + rrow) * DD;
            op[0 * 16 + m] = o0[r];
            op[1 * 16 + m] = o1[r];
            op[2 * 16 + m] = o2[r];
            op[3 * 16 + m] = o3[r];
        }
    }
}

// ---------------------------------------------------------------------------
// K4b: log-sum-exp combine of the KC partials; scatter into output rows.
//      Grid = BH*UU blocks of DD threads.
// ---------------------------------------------------------------------------
__global__ void k_attn_red(const float* __restrict__ Opart, const float* __restrict__ mpart,
                           const float* __restrict__ lpart, const int* __restrict__ topk,
                           float* __restrict__ out, int KC)
{
    int bh  = blockIdx.x / UU;
    int row = blockIdx.x % UU;
    int d   = threadIdx.x;

    float gm = -INFINITY;
    for (int c = 0; c < KC; ++c)
        gm = fmaxf(gm, mpart[((size_t)bh * KC + c) * UU + row]);

    float l = 0.f, acc = 0.f;
    for (int c = 0; c < KC; ++c) {
        size_t pc = (size_t)bh * KC + c;
        float w = __expf(mpart[pc * UU + row] - gm);
        l   += lpart[pc * UU + row] * w;
        acc += Opart[(pc * UU + row) * DD + d] * w;
    }
    out[((size_t)bh * LL + topk[bh * UU + row]) * DD + d] = acc / l;
}

// ---------------------------------------------------------------------------
extern "C" void kernel_launch(void* const* d_in, const int* in_sizes, int n_in,
                              void* d_out, int out_size, void* d_ws, size_t ws_size,
                              hipStream_t stream)
{
    const float* Q    = (const float*)d_in[0];
    const float* K    = (const float*)d_in[1];
    const float* V    = (const float*)d_in[2];
    const int*   samp = (const int*)d_in[3];
    float* out = (float*)d_out;

    const int BH = BB * HH;

    // fixed-size scratch
    const size_t szM        = (size_t)BH * LL * sizeof(float);     // 256 KB
    const size_t szTopk     = (size_t)BH * UU * sizeof(int);
    const size_t szMeanPart = (size_t)BH * MC * DD * sizeof(float);
    const size_t szMean     = (size_t)BH * DD * sizeof(float);
    const size_t fixed      = szM + szTopk + szMeanPart + szMean;

    auto splitBytes = [&](int kc) {
        return (size_t)BH * kc * UU * sizeof(float) * 2           // m, l
             + (size_t)BH * kc * UU * DD * sizeof(float);         // O
    };
    int KC = (ws_size >= fixed + splitBytes(8)) ? 8 : 1;

    char* ws = (char*)d_ws;
    float* M        = (float*)ws;
    int*   topk     = (int*)  (ws + szM);
    float* meanpart = (float*)(ws + szM + szTopk);
    float* mean     = (float*)(ws + szM + szTopk + szMeanPart);
    float* mpart    = (float*)(ws + fixed);
    float* lpart    = mpart + (size_t)BH * KC * UU;
    float* Opart    = lpart + (size_t)BH * KC * UU;

    k_metric   <<<(BH * LL + 255) / 256, 256, 0, stream>>>(Q, K, samp, M);
    k_topk     <<<BH, 256, 0, stream>>>(M, topk);
    k_mean_part<<<BH * MC, 256, 0, stream>>>(V, meanpart);
    k_mean_red <<<BH, DD, 0, stream>>>(meanpart, mean);
    {
        size_t total = (size_t)BH * LL * DD;
        k_fill<<<(unsigned)((total + 255) / 256), 256, 0, stream>>>(mean, out);
    }
    k_attn_part<<<BH * KC, 96, 0, stream>>>(Q, K, V, topk, Opart, mpart, lpart, KC);
    k_attn_red <<<BH * UU, DD, 0, stream>>>(Opart, mpart, lpart, topk, out, KC);
}
